// mhead_atten_42966852829549
// MI455X (gfx1250) — compile-verified, matrix operand order checked by reading, not run on hardware
//
#include <hip/hip_runtime.h>
#include <hip/hip_bf16.h>
#include <cstdint>

typedef __attribute__((ext_vector_type(16))) _Float16 v16h;
typedef __attribute__((ext_vector_type(8)))  _Float16 v8h;
typedef __attribute__((ext_vector_type(8)))  float    v8f;
typedef __attribute__((ext_vector_type(4)))  unsigned int u32x4;
typedef __attribute__((ext_vector_type(8)))  unsigned int u32x8;

#define NB    4
#define SEQ   2048
#define DK    1024
#define NH    16
#define DH    64
#define MROWS (NB * SEQ)   // 8192

// ---------------------------------------------------------------------------
// Fragment helpers (wave32 WMMA 16x16x32 f16)
// A/B fragment: lane (l&15) = row (A: M, B: N), (l>>4) selects K-subgroup.
// halves 0..7  <- p[0..7]   (K = kb .. kb+7)
// halves 8..15 <- p[16..23] (K = 16+kb ..)
// C/D layout: reg r of lane holds element (M = r + 8*(l>>4), N = l&15).
// ---------------------------------------------------------------------------
static __device__ __forceinline__ v16h load_frag16(const _Float16* p) {
  v8h lo = *(const v8h*)(p);
  v8h hi = *(const v8h*)(p + 16);
  v16h r;
#pragma unroll
  for (int i = 0; i < 8; ++i) { r[i] = lo[i]; r[8 + i] = hi[i]; }
  return r;
}

static __device__ __forceinline__ v8f wmma_f16(v16h a, v16h b, v8f c) {
  return __builtin_amdgcn_wmma_f32_16x16x32_f16(false, a, false, b,
                                                (short)0, c, false, false);
}

static __device__ __forceinline__ v8f v8f_zero() {
  v8f z = {0.f, 0.f, 0.f, 0.f, 0.f, 0.f, 0.f, 0.f};
  return z;
}

// ---------------------------------------------------------------------------
// 16-lane row max via v_permlane16_b32 butterflies (pure VALU, no LDS).
// Lane i combines with lane i^k inside its 16-lane row; both wave32 halves
// reduce independently (matches the WMMA C/D layout's per-half columns).
// ---------------------------------------------------------------------------
static __device__ __forceinline__ float pl16_max(float v, unsigned s0,
                                                 unsigned s1) {
  const int iv = __float_as_int(v);
  const int sw = __builtin_amdgcn_permlane16(iv, iv, (int)s0, (int)s1,
                                             false, false);
  return fmaxf(v, __int_as_float(sw));
}
static __device__ __forceinline__ float rowmax16(float v) {
  v = pl16_max(v, 0x67452301u, 0xEFCDAB89u);  // xor 1
  v = pl16_max(v, 0x45670123u, 0xCDEF89ABu);  // xor 3
  v = pl16_max(v, 0x01234567u, 0x89ABCDEFu);  // xor 7
  v = pl16_max(v, 0xFEDCBA98u, 0x76543210u);  // xor 8
  return v;
}

// ---------------------------------------------------------------------------
// Tensor Data Mover: async 2D tile load global->LDS (CDNA5 TDM, D# per
// CDNA5 ISA). data_size=2B. LDS padding: after every
// 16 DWORDs (one 32-half row) insert 4 DWORDs -> padded row stride 40 halves.
// Values are wave-uniform; forced into SGPRs via readfirstlane for the "s"
// inline-asm constraints. Tracked by TENSORcnt.
// ---------------------------------------------------------------------------
static __device__ __forceinline__ void tdm_load_2d(
    const _Float16* gptr, unsigned lds_byte_addr,
    unsigned tensor_d0, unsigned tensor_d1, unsigned stride0,
    unsigned tile_d0, unsigned tile_d1) {
  const unsigned long long ga = (unsigned long long)(uintptr_t)gptr;
  u32x4 g0;
  g0[0] = 1u;                                          // count=1, user mode
  g0[1] = lds_byte_addr;                               // lds_addr
  g0[2] = (unsigned)(ga & 0xFFFFFFFFu);                // global_addr[31:0]
  g0[3] = (unsigned)((ga >> 32) & 0x1FFFFFFu)          // global_addr[56:32]
        | (2u << 30);                                  // type = 2 (image)
  u32x8 g1;
  g1[0] = (1u << 16)                                   // data_size = 2 bytes
        | (1u << 20)                                   // pad_enable
        | (3u << 22)                                   // pad_interval: 16 DW
        | (3u << 25);                                  // pad_amount:   4 DW
  g1[1] = (tensor_d0 & 0xFFFFu) << 16;                 // tensor_dim0 lo16
  g1[2] = ((tensor_d0 >> 16) & 0xFFFFu)                // tensor_dim0 hi16
        | ((tensor_d1 & 0xFFFFu) << 16);               // tensor_dim1 lo16
  g1[3] = ((tensor_d1 >> 16) & 0xFFFFu)                // tensor_dim1 hi16
        | ((tile_d0 & 0xFFFFu) << 16);                 // tile_dim0
  g1[4] = (tile_d1 & 0xFFFFu);                         // tile_dim1 (tile_dim2=0)
  g1[5] = stride0;                                     // tensor_dim0_stride lo32
  g1[6] = 0u;
  g1[7] = 0u;
#pragma unroll
  for (int i = 0; i < 4; ++i)
    g0[i] = (unsigned)__builtin_amdgcn_readfirstlane((int)g0[i]);
#pragma unroll
  for (int i = 0; i < 8; ++i)
    g1[i] = (unsigned)__builtin_amdgcn_readfirstlane((int)g1[i]);
  asm volatile("tensor_load_to_lds %0, %1" :: "s"(g0), "s"(g1) : "memory");
}

// ---------------------------------------------------------------------------
// fp32 -> fp16 conversions
// ---------------------------------------------------------------------------
__global__ void cvt_f32_to_f16(const float* __restrict__ in,
                               _Float16* __restrict__ out, int n) {
  int i = blockIdx.x * blockDim.x + threadIdx.x;
  if (i < n) out[i] = (_Float16)in[i];
}

// W[k][n] (KxN row-major fp32) -> Wt[n][k] fp16 (so B fragments read contiguous K)
__global__ void cvt_transpose_w(const float* __restrict__ W,
                                _Float16* __restrict__ Wt, int K, int N) {
  int i = blockIdx.x * blockDim.x + threadIdx.x;
  if (i < K * N) {
    int k = i / N, n = i - k * N;
    Wt[(size_t)n * K + k] = (_Float16)W[i];
  }
}

// ---------------------------------------------------------------------------
// WMMA GEMM: Out = (A[M,K] @ Bt[N,K]^T + bias) * out_scale
// 256 threads = 8 waves (4Mx2N); block tile 256x128; wave tile 64x64
// (4x4 WMMA accumulators). A/B staged through double-buffered LDS by the
// Tensor Data Mover (each wave DMAs its own 32-row A / 16-row B slice).
// MODE 0: f16 out, head layout [B,H,S,DH]
// MODE 1: f16 out, head-transposed [B,H,DH,S]   (for V)
// MODE 2: f32 out, row-major [M,DK]             (final projection)
// ---------------------------------------------------------------------------
template <int MODE>
__global__ __launch_bounds__(256)
void gemm_wmma(const _Float16* __restrict__ A, const _Float16* __restrict__ Bt,
               const float* __restrict__ bias, void* __restrict__ OutRaw,
               int K, float out_scale) {
  constexpr int LDT = 40;                       // padded row stride (halves)
  __shared__ _Float16 smem[2][(256 + 128) * LDT];   // 60 KB double-buffered

  const int lane = threadIdx.x & 31;
  const int wave = threadIdx.x >> 5;
  const int wm = wave & 3, wn = wave >> 2;
  const int mblk = blockIdx.x * 256;
  const int nblk = blockIdx.y * 128;
  const int mr = lane & 15;
  const int kb = (lane >> 4) * 8;

  v8f acc[4][4];
#pragma unroll
  for (int i = 0; i < 4; ++i)
#pragma unroll
    for (int j = 0; j < 4; ++j) acc[i][j] = v8f_zero();

  const unsigned lds0 = (unsigned)(uintptr_t)(&smem[0][0]);
  const unsigned bufBytes = (unsigned)((256 + 128) * LDT * 2);

  // stage k-panel `k0` into buffer `buf`: this wave's A slice (32 rows) and
  // B slice (16 rows), each row 32 halves -> TDM pads to 40-half stride.
  auto stage = [&](int buf, int k0) {
    tdm_load_2d(A + (size_t)(mblk + wave * 32) * K + k0,
                lds0 + (unsigned)buf * bufBytes + (unsigned)(wave * 32 * LDT * 2),
                (unsigned)K, (unsigned)MROWS, (unsigned)K, 32u, 32u);
    tdm_load_2d(Bt + (size_t)(nblk + wave * 16) * K + k0,
                lds0 + (unsigned)buf * bufBytes +
                    (unsigned)((256 + wave * 16) * LDT * 2),
                (unsigned)K, (unsigned)DK, (unsigned)K, 32u, 16u);
  };

  stage(0, 0);
  int buf = 0;
  for (int k0 = 0; k0 < K; k0 += 32) {
    if (k0 + 32 < K) {
      stage(buf ^ 1, k0 + 32);                     // prefetch next panel
      __builtin_amdgcn_s_wait_tensorcnt(2);        // current panel complete
    } else {
      __builtin_amdgcn_s_wait_tensorcnt(0);
    }
    __syncthreads();                               // all slices visible

    const _Float16* As = &smem[buf][0];
    const _Float16* Bs = &smem[buf][256 * LDT];
    v16h a[4], b[4];
#pragma unroll
    for (int i = 0; i < 4; ++i) {
      a[i] = load_frag16(As + (wm * 64 + i * 16 + mr) * LDT + kb);
      b[i] = load_frag16(Bs + (wn * 64 + i * 16 + mr) * LDT + kb);
    }
#pragma unroll
    for (int i = 0; i < 4; ++i)
#pragma unroll
      for (int j = 0; j < 4; ++j)
        acc[i][j] = wmma_f16(a[i], b[j], acc[i][j]);

    __syncthreads();                               // safe to overwrite buf
    buf ^= 1;
  }

  const int ncl = lane & 15;
  const int mhalf = (lane >> 4) * 8;
#pragma unroll
  for (int i = 0; i < 4; ++i) {
#pragma unroll
    for (int j = 0; j < 4; ++j) {
      const int ncol = nblk + wn * 64 + j * 16 + ncl;
      const float bv = bias[ncol];
#pragma unroll
      for (int r = 0; r < 8; ++r) {
        const int mrow = mblk + wm * 64 + i * 16 + mhalf + r;
        const float val = (acc[i][j][r] + bv) * out_scale;
        if (MODE == 2) {
          ((float*)OutRaw)[(size_t)mrow * DK + ncol] = val;
        } else {
          const int bi = mrow >> 11, s = mrow & (SEQ - 1);
          const int hh = ncol >> 6, d = ncol & (DH - 1);
          size_t idx;
          if (MODE == 0) idx = (((size_t)(bi * NH + hh) * SEQ) + s) * DH + d;
          else           idx = (((size_t)(bi * NH + hh) * DH) + d) * SEQ + s;
          ((_Float16*)OutRaw)[idx] = (_Float16)val;
        }
      }
    }
  }
}

// ---------------------------------------------------------------------------
// Flash attention: per block = one (b,h), 64 query rows (4 waves x 16 rows).
// Q pre-scaled by 1/sqrt(DH). K layout [B,H,S,DH]; V layout [B,H,DH,S].
// Per 32-key step: 4 WMMA for QK^T, online softmax (row max via VALU
// permlane16 butterflies; row SUM via an extra WMMA against an all-ones B
// fragment -> denominator accumulated in the matrix pipe), P reshaped via
// per-wave LDS tile, 4 WMMA for PV. Output f16 -> merged.
// ---------------------------------------------------------------------------
__global__ __launch_bounds__(128)
void attn_wmma(const _Float16* __restrict__ Qh, const _Float16* __restrict__ Kh,
               const _Float16* __restrict__ Vt, _Float16* __restrict__ Mg) {
  __shared__ _Float16 Pb[4][16][32];  // per-wave P tile (16 q rows x 32 keys)

  const int lane = threadIdx.x & 31;
  const int w = threadIdx.x >> 5;
  const int bh = blockIdx.y;            // b*NH + h
  const int b = bh >> 4, h = bh & (NH - 1);
  const int q0 = blockIdx.x * 64 + w * 16;
  const int mr = lane & 15;
  const int kb = (lane >> 4) * 8;
  const int mhalf = (lane >> 4) * 8;

  const _Float16* Qp = Qh + ((size_t)bh * SEQ + q0) * DH;
  const _Float16* Kp = Kh + (size_t)bh * SEQ * DH;
  const _Float16* Vp = Vt + (size_t)bh * DH * SEQ;

  v16h qa[2];
#pragma unroll
  for (int t = 0; t < 2; ++t)
    qa[t] = load_frag16(Qp + (size_t)mr * DH + t * 32 + kb);

  v16h onesf;
#pragma unroll
  for (int i = 0; i < 16; ++i) onesf[i] = (_Float16)1.0f;

  float mrun[8], alpha[8];
  v8f o[4];
  v8f lacc = v8f_zero();                // running softmax denominator (P @ 1)
#pragma unroll
  for (int r = 0; r < 8; ++r) mrun[r] = -1e30f;
#pragma unroll
  for (int j = 0; j < 4; ++j) o[j] = v8f_zero();

  for (int kc = 0; kc < SEQ; kc += 32) {
    // ---- scores: two 16x16 tiles (keys kc..kc+15, kc+16..kc+31) ----
    v8f sc[2];
#pragma unroll
    for (int t = 0; t < 2; ++t) {
      sc[t] = v8f_zero();
      const _Float16* kp = Kp + (size_t)(kc + t * 16 + mr) * DH + kb;
#pragma unroll
      for (int u = 0; u < 2; ++u)               // dh 0..31, 32..63
        sc[t] = wmma_f16(qa[u], load_frag16(kp + u * 32), sc[t]);
    }

    // ---- online softmax max + exp (row = r + mhalf) ----
#pragma unroll
    for (int r = 0; r < 8; ++r) {
      const float vmx = rowmax16(fmaxf(sc[0][r], sc[1][r]));
      const float nm = fmaxf(mrun[r], vmx);
      alpha[r] = __expf(mrun[r] - nm);
      mrun[r] = nm;
      const float p0 = __expf(sc[0][r] - nm);
      const float p1 = __expf(sc[1][r] - nm);
      Pb[w][mhalf + r][mr]      = (_Float16)p0;
      Pb[w][mhalf + r][16 + mr] = (_Float16)p1;
    }
    asm volatile("s_wait_dscnt 0" ::: "memory");

    // reload P as an A-fragment (16 rows x 32 keys)
    v16h pa;
#pragma unroll
    for (int i = 0; i < 8; ++i) {
      pa[i]     = Pb[w][mr][kb + i];
      pa[8 + i] = Pb[w][mr][16 + kb + i];
    }

    // ---- denominator: lacc = alpha*lacc + P@1 (one WMMA, no shuffles) ----
#pragma unroll
    for (int r = 0; r < 8; ++r) lacc[r] *= alpha[r];
    lacc = wmma_f16(pa, onesf, lacc);

    // ---- PV: rescale running O, then 4 WMMAs over dh chunks ----
#pragma unroll
    for (int j = 0; j < 4; ++j) {
#pragma unroll
      for (int r = 0; r < 8; ++r) o[j][r] *= alpha[r];
      const v16h vb = load_frag16(Vp + (size_t)(j * 16 + mr) * SEQ + kc + kb);
      o[j] = wmma_f16(pa, vb, o[j]);
    }
  }

  // ---- normalize + store merged [B,S,H*DH] f16 ----
#pragma unroll
  for (int j = 0; j < 4; ++j) {
#pragma unroll
    for (int r = 0; r < 8; ++r) {
      const int qrow = q0 + mhalf + r;
      const float val = o[j][r] / lacc[r];
      Mg[(size_t)(b * SEQ + qrow) * DK + h * DH + j * 16 + mr] = (_Float16)val;
    }
  }
}

// ---------------------------------------------------------------------------
// Host launcher
// ---------------------------------------------------------------------------
extern "C" void kernel_launch(void* const* d_in, const int* in_sizes, int n_in,
                              void* d_out, int out_size, void* d_ws, size_t ws_size,
                              hipStream_t stream) {
  (void)in_sizes; (void)n_in; (void)out_size; (void)ws_size;

  const float* q  = (const float*)d_in[0];
  const float* k  = (const float*)d_in[1];
  const float* v  = (const float*)d_in[2];
  const float* Wq = (const float*)d_in[3];
  const float* bq = (const float*)d_in[4];
  const float* Wk = (const float*)d_in[5];
  const float* bk = (const float*)d_in[6];
  const float* Wv = (const float*)d_in[7];
  const float* bv = (const float*)d_in[8];
  const float* Wu = (const float*)d_in[9];
  const float* bu = (const float*)d_in[10];

  char* ws = (char*)d_ws;
  _Float16* Wt  = (_Float16*)(ws);                          //  2 MB
  _Float16* X16 = (_Float16*)(ws + ( 2ull << 20));          // 16 MB (x / merged)
  _Float16* Qh  = (_Float16*)(ws + (18ull << 20));          // 16 MB
  _Float16* Kh  = (_Float16*)(ws + (34ull << 20));          // 16 MB
  _Float16* Vt  = (_Float16*)(ws + (50ull << 20));          // 16 MB -> 66 MB total

  const int NW = DK * DK;
  const int NX = MROWS * DK;
  dim3 cblk(256);
  dim3 g_w((NW + 255) / 256);
  dim3 g_x((NX + 255) / 256);
  dim3 g_gemm(MROWS / 256, DK / 128);
  dim3 g_attn(SEQ / 64, NB * NH);

  const float qscale = 0.125f;  // 1/sqrt(DH) folded into Q

  // Q projection
  cvt_transpose_w<<<g_w, cblk, 0, stream>>>(Wq, Wt, DK, DK);
  cvt_f32_to_f16<<<g_x, cblk, 0, stream>>>(q, X16, NX);
  gemm_wmma<0><<<g_gemm, dim3(256), 0, stream>>>(X16, Wt, bq, Qh, DK, qscale);
  // K projection
  cvt_transpose_w<<<g_w, cblk, 0, stream>>>(Wk, Wt, DK, DK);
  cvt_f32_to_f16<<<g_x, cblk, 0, stream>>>(k, X16, NX);
  gemm_wmma<0><<<g_gemm, dim3(256), 0, stream>>>(X16, Wt, bk, Kh, DK, 1.0f);
  // V projection (stored [B,H,DH,S] for the PV B-fragment path)
  cvt_transpose_w<<<g_w, cblk, 0, stream>>>(Wv, Wt, DK, DK);
  cvt_f32_to_f16<<<g_x, cblk, 0, stream>>>(v, X16, NX);
  gemm_wmma<1><<<g_gemm, dim3(256), 0, stream>>>(X16, Wt, bv, Vt, DK, 1.0f);
  // Flash attention -> merged (reuses X16)
  attn_wmma<<<g_attn, dim3(128), 0, stream>>>(Qh, Kh, Vt, X16);
  // Output projection -> fp32 d_out
  cvt_transpose_w<<<g_w, cblk, 0, stream>>>(Wu, Wt, DK, DK);
  gemm_wmma<2><<<g_gemm, dim3(256), 0, stream>>>(X16, Wt, bu, d_out, DK, 1.0f);
}
